// MaskingExtractor_4913442587228
// MI455X (gfx1250) — matrix operand order repaired and checked by reading, not compiled, not run on hardware
//
#include <hip/hip_runtime.h>
#include <hip/hip_bf16.h>

// ---------------------------------------------------------------------------
// MaskingExtractor pipeline for gfx1250 (MI455X), f16 WMMA path with
// TDM (tensor_load_to_lds) weight staging + async-to-LDS activation staging,
// double-buffered on TENSORcnt/ASYNCcnt.
//   B=32768, D_IN=768, GEO=11, H=512, OUT=512, L=6, NUM_MASKED=230
// ---------------------------------------------------------------------------

typedef __attribute__((ext_vector_type(16))) _Float16 v16h;
typedef __attribute__((ext_vector_type(8)))  _Float16 v8h;
typedef __attribute__((ext_vector_type(8)))  float    v8f;
typedef __attribute__((ext_vector_type(4)))  unsigned int v4u;
typedef __attribute__((ext_vector_type(8)))  int      v8i;
typedef __attribute__((ext_vector_type(4)))  int      v4i;

#define BQ 32768
#define DIN 768
#define HID 512
#define NMASK 230

#if __has_builtin(__builtin_amdgcn_tensor_load_to_lds)
#define HAVE_TDM 1
#else
#define HAVE_TDM 0
#endif

// Raw 32-bit LDS byte offset of a __shared__ object (async/TDM dest operand).
__device__ __forceinline__ unsigned ldsAddr(const void* p) {
    return (unsigned)(unsigned long long)(const __attribute__((address_space(3))) char*)p;
}

#if HAVE_TDM
// One 2D TDM descriptor: tile 128 rows x 32 f16, row stride K elements,
// LDS rows padded 64B data + 16B pad (pad_interval code 3 = 16 DWORDs,
// pad_amount code 3 = 4 DWORDs) -> 80B pitch == LDSP halves.
__device__ __forceinline__ void tdm_load_tile(unsigned ldsByte, const void* gp, int K) {
    const unsigned long long ga = (unsigned long long)gp;
    v4u g0;
    g0.x = 1u;                                          // count=1, user mode
    g0.y = ldsByte;                                     // lds_addr
    g0.z = (unsigned)ga;                                // global_addr[31:0]
    g0.w = (unsigned)((ga >> 32) & 0x1FFFFFFu) | (2u << 30);  // [56:32] | type=2
    v8i g1;
    g1[0] = (int)((1u << 16) | (1u << 20) | (3u << 22) | (3u << 25));
    //            data_size=2B  pad_enable  pad_interval=16dw  pad_amount=4dw
    g1[1] = (int)((unsigned)K << 16);                   // tensor_dim0[15:0]
    g1[2] = (int)(128u << 16);                          // tensor_dim1 = 128
    g1[3] = (int)(32u << 16);                           // tile_dim0 = 32
    g1[4] = 128;                                        // tile_dim1=128, tile_dim2=0
    g1[5] = K;                                          // tensor_dim0_stride lo
    g1[6] = 0;
    g1[7] = 0;
    const v4i z4 = {0, 0, 0, 0};
#if defined(__clang_major__) && __clang_major__ >= 23
    const v8i z8 = {0, 0, 0, 0, 0, 0, 0, 0};
    __builtin_amdgcn_tensor_load_to_lds(g0, g1, z4, z4, z8, 0);
#else
    __builtin_amdgcn_tensor_load_to_lds(g0, g1, z4, z4, 0);
#endif
}
#endif

// ---------------------------------------------------------------------------
// 1) Per-row threshold: bisection for the NMASK-th smallest u in [0,1).
// ---------------------------------------------------------------------------
__global__ void thresh_kernel(const float* __restrict__ U, float* __restrict__ thr) {
    const int row = blockIdx.x;
    const int t   = threadIdx.x;
    const float* u = U + (size_t)row * DIN;
    const float u0 = u[t];
    const float u1 = u[t + 256];
    const float u2 = u[t + 512];
    __shared__ int sred[256];
    float lo = 0.0f, hi = 1.0f;
    for (int it = 0; it < 40; ++it) {
        const float mid = 0.5f * (lo + hi);
        int c = (u0 < mid) + (u1 < mid) + (u2 < mid);
        sred[t] = c;
        __syncthreads();
        for (int s = 128; s > 0; s >>= 1) {
            if (t < s) sred[t] += sred[t + s];
            __syncthreads();
        }
        const int total = sred[0];
        __syncthreads();
        if (total >= NMASK) hi = mid; else lo = mid;
    }
    if (t == 0) thr[row] = lo;
}

// ---------------------------------------------------------------------------
// 2) Weight transpose + cast: src f32 [K,N] row-major -> dst f16 [N,K].
// ---------------------------------------------------------------------------
__global__ void transpose_cast_kernel(const float* __restrict__ src,
                                      _Float16* __restrict__ dst,
                                      int K, int N) {
    const int idx = blockIdx.x * blockDim.x + threadIdx.x;
    if (idx >= K * N) return;
    const int k = idx / N;
    const int n = idx - k * N;
    dst[(size_t)n * K + k] = (_Float16)src[idx];
}

// ---------------------------------------------------------------------------
// 3) Geo projection: g[b,n] = geo[b,:] @ geoW[:,n] + geo_b[n]   (GEO=11)
// ---------------------------------------------------------------------------
__global__ void geo_kernel(const float* __restrict__ geo,
                           const float* __restrict__ geoW,
                           const float* __restrict__ geob,
                           _Float16* __restrict__ g) {
    const int idx = blockIdx.x * blockDim.x + threadIdx.x;
    if (idx >= BQ * HID) return;
    const int b = idx >> 9;
    const int n = idx & 511;
    float s = geob[n];
#pragma unroll
    for (int j = 0; j < 11; ++j)
        s += geo[(size_t)b * 11 + j] * geoW[j * HID + n];
    g[idx] = (_Float16)s;
}

// ---------------------------------------------------------------------------
// 4) 3-channel, 5-tap 1D conv (SAME zero pad) collapsing [B,3,512]->[B,512].
// ---------------------------------------------------------------------------
__global__ void conv_kernel(const _Float16* __restrict__ pi,
                            const _Float16* __restrict__ pj,
                            const _Float16* __restrict__ g,
                            const float* __restrict__ cw,   // [1,3,5]
                            const float* __restrict__ cb,   // [1]
                            _Float16* __restrict__ out) {
    const int idx = blockIdx.x * blockDim.x + threadIdx.x;
    if (idx >= BQ * HID) return;
    const int b = idx >> 9;
    const int h = idx & 511;
    const size_t rb = (size_t)b * HID;
    float s = cb[0];
#pragma unroll
    for (int k = 0; k < 5; ++k) {
        const int p = h + k - 2;
        if (p >= 0 && p < HID) {
            s += cw[k]      * (float)pi[rb + p];
            s += cw[5 + k]  * (float)pj[rb + p];
            s += cw[10 + k] * (float)g[rb + p];
        }
    }
    out[idx] = (_Float16)s;
}

// ---------------------------------------------------------------------------
// 5) WMMA GEMM: O = act(A[M,K] @ W[K,N] + bias [+ Res])
//    B (weight) tile: TDM tensor_load_to_lds (wave 0 issues; TENSORcnt).
//    A tile: f16 via global_load_async_to_lds_b128 (ASYNCcnt), or masked f32
//    (u>thr ? x : 0) register-prefetch + ds_store.
//    256 thr = 8 wave32; block tile 128x128; K-step 32; wave tile 32x64.
// ---------------------------------------------------------------------------
#define LDSP 40   // row pitch in halves (32 data + 8 pad = 80B); 16B aligned

template <int AMODE, bool RELU, bool RES, bool O16, bool O32>
__global__ __launch_bounds__(256)
void gemm_wmma(const void* __restrict__ A,
               const float* __restrict__ U,
               const float* __restrict__ thr,
               const _Float16* __restrict__ WT,
               const float* __restrict__ bias,
               const _Float16* __restrict__ Rsd,
               _Float16* __restrict__ Of16,
               float* __restrict__ Of32,
               int M, int N, int K) {
    __shared__ _Float16 sA[2][128 * LDSP];
    __shared__ _Float16 sB[2][128 * LDSP];

    const int tid    = threadIdx.x;
    const int lane   = tid & 31;
    const int wave   = tid >> 5;
    const int wm     = wave >> 1;        // 0..3 -> 32-row strips
    const int wn     = wave & 1;         // 0..1 -> 64-col strips
    const int lrow   = lane & 15;
    const int lhi    = lane >> 4;
    const int blockN = blockIdx.x * 128;
    const int blockM = blockIdx.y * 128;
    const bool isW0  = (__builtin_amdgcn_readfirstlane(wave) == 0);  // uniform

    // tile-loader mapping: thread -> (row, 16-half chunk)
    const int lr = tid >> 1;             // 0..127
    const int lc = (tid & 1) * 16;       // 0 or 16

    const unsigned aSlot0 = ldsAddr(&sA[0][lr * LDSP + lc]);
    const unsigned aSlot1 = ldsAddr(&sA[1][lr * LDSP + lc]);
    const unsigned bBase0 = ldsAddr(&sB[0][0]);
    const unsigned bBase1 = ldsAddr(&sB[1][0]);
    const unsigned aGlobBase = ((unsigned)(blockM + lr) * (unsigned)K + (unsigned)lc) * 2u;

    // ---- B tile: one TDM descriptor per stage (wave 0 only) ----
    auto issueB = [&](int k0, int bufSel) {
#if HAVE_TDM
        tdm_load_tile(bufSel ? bBase1 : bBase0,
                      WT + (size_t)blockN * K + k0, K);
#else
        const unsigned l = (bufSel ? bBase1 : bBase0) + (unsigned)(lr * LDSP + lc) * 2u;
        const unsigned g = ((unsigned)(blockN + lr) * (unsigned)K + (unsigned)(k0 + lc)) * 2u;
        asm volatile(
            "global_load_async_to_lds_b128 %0, %1, %2\n\t"
            "global_load_async_to_lds_b128 %0, %1, %2 offset:16"
            :: "v"(l), "v"(g), "s"(WT) : "memory");
#endif
    };
    // ---- A tile (f16 mode): per-lane async-to-LDS ----
    auto issueA16 = [&](int k0, int bufSel) {
        const unsigned l = bufSel ? aSlot1 : aSlot0;
        const unsigned g = aGlobBase + (unsigned)k0 * 2u;
        asm volatile(
            "global_load_async_to_lds_b128 %0, %1, %2\n\t"
            "global_load_async_to_lds_b128 %0, %1, %2 offset:16"
            :: "v"(l), "v"(g), "s"((const _Float16*)A) : "memory");
    };

    // ---- masked-A register prefetch state (AMODE 1) ----
    float4 px[4], pu[4];
    float th = 0.0f;
    auto fetchMasked = [&](int k0) {
        const float* Af = (const float*)A;
        const int gr = blockM + lr;
        const float* xrow = Af + (size_t)gr * K + k0 + lc;
        const float* urow = U  + (size_t)gr * K + k0 + lc;
#pragma unroll
        for (int q = 0; q < 4; ++q) {
            px[q] = ((const float4*)xrow)[q];
            pu[q] = ((const float4*)urow)[q];
        }
    };
    auto storeMasked = [&](int bufSel) {
        __attribute__((aligned(16))) _Float16 tmp[16];
#pragma unroll
        for (int q = 0; q < 4; ++q) {
            tmp[q * 4 + 0] = (pu[q].x > th) ? (_Float16)px[q].x : (_Float16)0.0f;
            tmp[q * 4 + 1] = (pu[q].y > th) ? (_Float16)px[q].y : (_Float16)0.0f;
            tmp[q * 4 + 2] = (pu[q].z > th) ? (_Float16)px[q].z : (_Float16)0.0f;
            tmp[q * 4 + 3] = (pu[q].w > th) ? (_Float16)px[q].w : (_Float16)0.0f;
        }
        _Float16* dst = &sA[bufSel][lr * LDSP + lc];
        *(uint4*)dst       = *(const uint4*)&tmp[0];
        *(uint4*)(dst + 8) = *(const uint4*)&tmp[8];
    };

    v8f acc[2][4];
#pragma unroll
    for (int i = 0; i < 2; ++i)
#pragma unroll
        for (int j = 0; j < 4; ++j) acc[i][j] = (v8f){};

    union V16U { v16h v; v8h h[2]; };

    // -------- prologue: stage 0 in flight --------
    if (AMODE == 1) {
        th = thr[blockM + lr];
        fetchMasked(0);
        storeMasked(0);
    } else {
        issueA16(0, 0);
    }
    if (isW0) issueB(0, 0);

    int buf = 0;
    for (int k0 = 0; k0 < K; k0 += 32) {
        const bool last = (k0 + 32 >= K);
        const int nbuf = buf ^ 1;

        if (AMODE == 1 && !last) fetchMasked(k0 + 32);   // long-latency, pre-barrier

        __syncthreads();               // (A) everyone done computing stage k-1

        if (!last) {                   // stage k+1 into the alternate buffer
            if (isW0) issueB(k0 + 32, nbuf);
            if (AMODE == 0) issueA16(k0 + 32, nbuf);
        }
        // Stage k's loads complete; stage k+1's stay outstanding.
        if (AMODE == 0) {
            if (!last) asm volatile("s_wait_asynccnt 2" ::: "memory");
            else       asm volatile("s_wait_asynccnt 0" ::: "memory");
        }
        if (isW0) {
#if HAVE_TDM
            if (!last) __builtin_amdgcn_s_wait_tensorcnt(1);
            else       __builtin_amdgcn_s_wait_tensorcnt(0);
#else
            if (!last) asm volatile("s_wait_asynccnt 2" ::: "memory");
            else       asm volatile("s_wait_asynccnt 0" ::: "memory");
#endif
        }

        __syncthreads();               // (B) stage k tile visible to all waves

        // ---- fragments (ISA 7.12.2 layouts) + 8 WMMAs ----
        const _Float16* cA = sA[buf];
        const _Float16* cB = sB[buf];
        v16h af[2];
#pragma unroll
        for (int mt = 0; mt < 2; ++mt) {
            const int rA = wm * 32 + mt * 16 + lrow;
            const int kA = lhi * 8;                       // lanes16-31 start K=8
            V16U a;
            a.h[0] = *(const v8h*)&cA[rA * LDSP + kA];       // K = kA..kA+7
            a.h[1] = *(const v8h*)&cA[rA * LDSP + kA + 16];  // K = kA+16..kA+23
            af[mt] = a.v;
        }
        v16h bfr[4];
#pragma unroll
        for (int nt = 0; nt < 4; ++nt) {
            const int nB = wn * 64 + nt * 16 + lrow;
            const int kB = lhi * 8;                       // lanes16-31 start K=8
            V16U b;
            b.h[0] = *(const v8h*)&cB[nB * LDSP + kB];
            b.h[1] = *(const v8h*)&cB[nB * LDSP + kB + 16];
            bfr[nt] = b.v;
        }
#pragma unroll
        for (int mt = 0; mt < 2; ++mt)
#pragma unroll
            for (int nt = 0; nt < 4; ++nt)
                acc[mt][nt] = __builtin_amdgcn_wmma_f32_16x16x32_f16(
                    false, af[mt], false, bfr[nt], (short)0, acc[mt][nt],
                    false, false);

        // masked stage k+1 ds_store after compute; published by next barrier
        if (AMODE == 1 && !last) storeMasked(nbuf);

        buf = nbuf;
    }

    // ---- epilogue: bias, (+Res), (ReLU), store f16/f32 ----
#pragma unroll
    for (int mt = 0; mt < 2; ++mt) {
#pragma unroll
        for (int nt = 0; nt < 4; ++nt) {
            const int rowBase = blockM + wm * 32 + mt * 16 + lhi * 8;
            const int colG    = blockN + wn * 64 + nt * 16 + lrow;
            const float bv = bias[colG];
            const v8f a = acc[mt][nt];
#pragma unroll
            for (int v = 0; v < 8; ++v) {
                const size_t off = (size_t)(rowBase + v) * N + colG;
                float val = a[v] + bv;
                if (RES)  val += (float)Rsd[off];
                if (RELU) val = val > 0.0f ? val : 0.0f;
                if (O16)  Of16[off] = (_Float16)val;
                if (O32)  Of32[off] = val;
            }
        }
    }
}

// ---------------------------------------------------------------------------
// Host-side orchestration
// ---------------------------------------------------------------------------
extern "C" void kernel_launch(void* const* d_in, const int* in_sizes, int n_in,
                              void* d_out, int out_size, void* d_ws, size_t ws_size,
                              hipStream_t stream) {
    const float* x_i   = (const float*)d_in[0];
    const float* x_j   = (const float*)d_in[1];
    const float* geo   = (const float*)d_in[2];
    const float* u_i   = (const float*)d_in[3];
    const float* u_j   = (const float*)d_in[4];
    const float* objW  = (const float*)d_in[5];
    const float* objb  = (const float*)d_in[6];
    const float* geoW  = (const float*)d_in[7];
    const float* geob  = (const float*)d_in[8];
    const float* convW = (const float*)d_in[9];
    const float* convb = (const float*)d_in[10];
    const float* rW1   = (const float*)d_in[11];
    const float* rb1   = (const float*)d_in[12];
    const float* rW2   = (const float*)d_in[13];
    const float* rb2   = (const float*)d_in[14];
    const float* fcW   = (const float*)d_in[15];
    const float* fcb   = (const float*)d_in[16];
    float* out = (float*)d_out;

    // workspace layout (bytes)
    char* ws = (char*)d_ws;
    float*     thrI  = (float*)(ws + 0);                       // 128 KB
    float*     thrJ  = (float*)(ws + 131072);                  // 128 KB
    _Float16*  objWT = (_Float16*)(ws + 262144);               // 512x768 f16
    _Float16*  W1T   = (_Float16*)(ws + 1048576);              // 6x512x512 f16
    _Float16*  W2T   = (_Float16*)(ws + 4194304);              // 6x512x512 f16
    _Float16*  fcWT  = (_Float16*)(ws + 7340032);              // 512x512 f16
    _Float16*  buf0  = (_Float16*)(ws + 7864320);              // [B,512] f16
    _Float16*  buf1  = (_Float16*)(ws + 41418752ULL);
    _Float16*  buf2  = (_Float16*)(ws + 74973184ULL);
    _Float16*  buf3  = (_Float16*)(ws + 108527616ULL);

    // 1) masking thresholds
    thresh_kernel<<<BQ, 256, 0, stream>>>(u_i, thrI);
    thresh_kernel<<<BQ, 256, 0, stream>>>(u_j, thrJ);

    // 2) weight transpose + f16 cast
    {
        int n = DIN * HID;
        transpose_cast_kernel<<<(n + 255) / 256, 256, 0, stream>>>(objW, objWT, DIN, HID);
        n = HID * HID;
        for (int l = 0; l < 6; ++l) {
            transpose_cast_kernel<<<(n + 255) / 256, 256, 0, stream>>>(
                rW1 + (size_t)l * n, W1T + (size_t)l * n, HID, HID);
            transpose_cast_kernel<<<(n + 255) / 256, 256, 0, stream>>>(
                rW2 + (size_t)l * n, W2T + (size_t)l * n, HID, HID);
        }
        transpose_cast_kernel<<<(n + 255) / 256, 256, 0, stream>>>(fcW, fcWT, HID, HID);
    }

    const dim3 gemmGrid(HID / 128, BQ / 128);   // (4, 256)

    // 3) p_i / p_j: masked x @ obj_W + obj_b   (fused mask, f32->f16 A path)
    gemm_wmma<1, false, false, true, false><<<gemmGrid, 256, 0, stream>>>(
        x_i, u_i, thrI, objWT, objb, nullptr, buf0, nullptr, BQ, HID, DIN);
    gemm_wmma<1, false, false, true, false><<<gemmGrid, 256, 0, stream>>>(
        x_j, u_j, thrJ, objWT, objb, nullptr, buf1, nullptr, BQ, HID, DIN);

    // 4) geo projection -> buf2
    {
        const int n = BQ * HID;
        geo_kernel<<<(n + 255) / 256, 256, 0, stream>>>(geo, geoW, geob, buf2);
        // 5) conv over the 3 stacked channels -> buf3 (= e_ij = initial x)
        conv_kernel<<<(n + 255) / 256, 256, 0, stream>>>(buf0, buf1, buf2, convW, convb, buf3);
    }

    // 6) residual blocks: x = relu(x + fc2(relu(fc1(x))))
    _Float16* x  = buf3;
    _Float16* f1 = buf0;
    _Float16* f2 = buf1;
    for (int l = 0; l < 6; ++l) {
        const size_t wo = (size_t)l * HID * HID;
        // h = relu(x @ W1 + b1) -> f1
        gemm_wmma<0, true, false, true, false><<<gemmGrid, 256, 0, stream>>>(
            x, nullptr, nullptr, W1T + wo, rb1 + (size_t)l * HID, nullptr,
            f1, nullptr, BQ, HID, HID);
        // x' = relu(x + h @ W2 + b2) -> f2
        gemm_wmma<0, true, true, true, false><<<gemmGrid, 256, 0, stream>>>(
            f1, nullptr, nullptr, W2T + wo, rb2 + (size_t)l * HID, x,
            f2, nullptr, BQ, HID, HID);
        _Float16* t = x; x = f2; f2 = f1; f1 = t;
    }

    // 7) final projection, f32 straight to d_out
    gemm_wmma<0, false, false, false, true><<<gemmGrid, 256, 0, stream>>>(
        x, nullptr, nullptr, fcWT, fcb, nullptr, nullptr, out, BQ, HID, HID);
}
